// GraphAttentionLayer_33612414058785
// MI455X (gfx1250) — matrix operand order repaired and checked by reading, not compiled, not run on hardware
//
#include <hip/hip_runtime.h>
#include <hip/hip_bf16.h>

// ---------------------------------------------------------------------------
// GraphAttentionLayer fused kernels for gfx1250 (MI455X, wave32, WMMA, TDM)
//   B=8, N=2048, D=256
//   k1: bf16 conversions (+ w_map pre-scale), fused [Watt|Wres] weight build,
//       BN stats zero
//   k2: LDS-tiled transpose x -> xT[b][d][n] bf16 (feeds P@V B-fragments)
//   k3: flash-attention (QK^T -> online softmax -> @V) fused with the
//       [agg|x] @ [Watt|Wres]^T projection; BN partials via f32 atomics.
//       K-tiles double-buffered in LDS via the Tensor Data Mover (async DMA,
//       TENSORcnt), QK^T B-frags from LDS, P@V B-frags from L2-resident xT.
//   k4: BatchNorm (training stats) + SELU, memory bound
// ---------------------------------------------------------------------------

typedef __attribute__((ext_vector_type(16))) __bf16 v16bf;
typedef __attribute__((ext_vector_type(8)))  float  v8f;
typedef __attribute__((ext_vector_type(4)))  unsigned int v4u;
typedef __attribute__((ext_vector_type(8)))  int    v8i;
typedef __attribute__((ext_vector_type(4)))  int    v4i;

namespace {
constexpr int Bc = 8;
constexpr int Nc = 2048;
constexpr int Dc = 256;
constexpr int ROWS_PER_WG = 128;   // 8 waves x 16 rows
constexpr int JT = 32;             // key tile per main-loop iteration

// LDS layout (bytes) for the attention kernel
constexpr int KT_BUF_BYTES = JT * Dc * 2;                 // 16384 per buffer
constexpr int PSTG_OFF     = 2 * KT_BUF_BYTES;            // 32768 (8 waves x 1KB)
constexpr int ESTG_OFF     = PSTG_OFF + 8 * 1024;         // 40960
constexpr int SMEM_BYTES   = ESTG_OFF + 128 * 512;        // 106496 (< 320KB/WG)
}

union Frag {
  v16bf bf;
  int4  i4[2];
};

__device__ __forceinline__ unsigned short f2bf(float f) {
  unsigned int u = __float_as_uint(f);
  u += 0x7FFFu + ((u >> 16) & 1u);      // round-to-nearest-even
  return (unsigned short)(u >> 16);
}

__device__ __forceinline__ v8f zero8() {
  v8f z;
#pragma unroll
  for (int i = 0; i < 8; ++i) z[i] = 0.0f;
  return z;
}

// ---------------------------------------------------------------------------
// Kernel 1: prep — bf16 copies, w_map pre-scale, fused transposed weights,
// zero BN stats.
// ---------------------------------------------------------------------------
__global__ void gat_prep_kernel(const float* __restrict__ x,
                                const float* __restrict__ w_map,
                                const float* __restrict__ w_att,
                                const float* __restrict__ w_res,
                                unsigned short* __restrict__ xbf,
                                unsigned short* __restrict__ qbf,
                                unsigned short* __restrict__ wcat,
                                float* __restrict__ stats) {
  size_t idx = (size_t)blockIdx.x * blockDim.x + threadIdx.x;
  size_t total = (size_t)Bc * Nc * Dc;
  if (idx < total) {
    float v = x[idx];
    xbf[idx] = f2bf(v);
    qbf[idx] = f2bf(v * w_map[idx & (Dc - 1)]);
  }
  if (idx < (size_t)Dc * 512) {               // wcat[n][k], k<256: Watt, k>=256: Wres
    int n = (int)(idx >> 9);
    int k = (int)(idx & 511);
    float w = (k < Dc) ? w_att[n * Dc + k] : w_res[n * Dc + (k - Dc)];
    wcat[idx] = f2bf(w);
  }
  if (idx < 512) stats[idx] = 0.0f;
}

// ---------------------------------------------------------------------------
// Kernel 2: LDS-tiled transpose  x[b][n][d] (f32) -> xT[b][d][n] (bf16).
// ---------------------------------------------------------------------------
__global__ void gat_transpose_kernel(const float* __restrict__ x,
                                     unsigned short* __restrict__ xT) {
  __shared__ unsigned short tile[32][33];    // +1 pad: conflict-free transpose
  const int tilesPerBatch = (Nc / 32) * (Dc / 32);
  int bIdx = blockIdx.x / tilesPerBatch;
  int rem  = blockIdx.x % tilesPerBatch;
  int n0 = (rem / (Dc / 32)) * 32;
  int d0 = (rem % (Dc / 32)) * 32;
  int r = threadIdx.x >> 5;                  // 0..7
  int c = threadIdx.x & 31;
#pragma unroll
  for (int q = 0; q < 4; ++q) {
    int row = r + q * 8;
    tile[row][c] = f2bf(x[((size_t)bIdx * Nc + n0 + row) * Dc + d0 + c]);
  }
  __syncthreads();
#pragma unroll
  for (int q = 0; q < 4; ++q) {
    int row = r + q * 8;
    xT[((size_t)bIdx * Dc + d0 + row) * Nc + n0 + c] = tile[c][row];
  }
}

// ---------------------------------------------------------------------------
// Kernel 3: fused flash attention + projection.
// grid.x = B * (N/128). 256 threads = 8 waves; each wave owns 16 query rows.
// ---------------------------------------------------------------------------
__global__ __launch_bounds__(256, 1)
void gat_attn_kernel(const unsigned short* __restrict__ xbf,
                     const unsigned short* __restrict__ qbf,
                     const unsigned short* __restrict__ xT,
                     const unsigned short* __restrict__ wcat,
                     float* __restrict__ out_pre,
                     float* __restrict__ stats) {
  extern __shared__ char smem[];
  char* ktbuf = smem;                         // 2 x [32][256] bf16 (double buffer)
  char* pstg  = smem + PSTG_OFF;              // 8 waves x [16][32] bf16
  char* estg  = smem + ESTG_OFF;              // [128][256] bf16 epilogue stage

  const int tid  = threadIdx.x;
  const int wave = tid >> 5;
  const int lane = tid & 31;
  const int l16  = lane & 15;
  const int lh   = lane >> 4;                 // lane half (row offset 0/8 in C frags)
  const int asel = (lane < 16) ? 0 : 32;      // A-fragment K-half byte offset
  const int bsel = (lane < 16) ? 0 : 32;      // B-fragment K-half byte offset

  const int bIdx = blockIdx.x / (Nc / ROWS_PER_WG);
  const int i0   = (blockIdx.x % (Nc / ROWS_PER_WG)) * ROWS_PER_WG;

  // ---- Tensor Data Mover: DMA one 32x256 bf16 K-tile into an LDS buffer.
  // D# group0: count=1 user descriptor, lds_addr, 57-bit global_addr, type=2.
  // D# group1: data_size=2B, tensor_dim0=256, tensor_dim1=2048,
  //            tile_dim0=256, tile_dim1=32, tensor_dim0_stride=256.
  const size_t xbBase = (size_t)bIdx * Nc * Dc;   // element offset of batch
  auto issue_tile_load = [&](int tile, int bufsel) {
    unsigned long long ga = (unsigned long long)(uintptr_t)
        (xbf + xbBase + (size_t)tile * JT * Dc);
    v4u g0;
    g0[0] = 1u;                                          // count=1
    g0[1] = (unsigned)(bufsel * KT_BUF_BYTES);           // lds_addr (bytes)
    g0[2] = (unsigned)(ga & 0xFFFFFFFFu);                // global_addr[31:0]
    g0[3] = (unsigned)((ga >> 32) & 0x1FFFFFFu) | (2u << 30);  // [56:32] | type
    v8i g1;
    g1[0] = (1 << 16);                                   // data_size = 2 bytes
    g1[1] = (int)((unsigned)Dc << 16);                   // tensor_dim0 lo16
    g1[2] = (int)(((unsigned)Nc & 0xFFFFu) << 16);       // tensor_dim1 lo16
    g1[3] = (int)((unsigned)Dc << 16);                   // tile_dim0 = 256
    g1[4] = JT;                                          // tile_dim1 = 32 rows
    g1[5] = Dc;                                          // dim0_stride = 256
    g1[6] = 0;
    g1[7] = 0;
    v4i gz4;
    gz4[0] = gz4[1] = gz4[2] = gz4[3] = 0;               // 2-D: groups 2/3 unused
    v8i gz8;
#pragma unroll
    for (int i = 0; i < 8; ++i) gz8[i] = 0;
    __builtin_amdgcn_tensor_load_to_lds(g0, g1, gz4, gz4, gz8, 0);
  };

  // ---- resident Q fragments (16 rows x 256 K), A layout with k = 16*half + s, from global
  Frag qa[8];
  {
    const char* base =
        (const char*)(qbf + ((size_t)bIdx * Nc + i0 + wave * 16) * Dc) +
        l16 * 512 + asel;
#pragma unroll
    for (int kk = 0; kk < 8; ++kk) {
      qa[kk].i4[0] = *(const int4*)(base + kk * 64);
      qa[kk].i4[1] = *(const int4*)(base + kk * 64 + 16);
    }
  }

  // ---- flash state: acc[16 d-tiles], running max/sum as half-wave rows
  v8f acc[16];
#pragma unroll
  for (int i = 0; i < 16; ++i) acc[i] = zero8();
  float mrow[8], lsum[8];
#pragma unroll
  for (int r = 0; r < 8; ++r) { mrow[r] = -1e30f; lsum[r] = 0.0f; }

  unsigned short* pw = (unsigned short*)(pstg + wave * 1024);

  // preload K-tile 0 into buffer 0 via TDM
  if (wave == 0) {
    issue_tile_load(0, 0);
    __builtin_amdgcn_s_wait_tensorcnt(0);
  }
  __syncthreads();

  constexpr int NT_TILES = Nc / JT;           // 64
#pragma unroll 1
  for (int t = 0; t < NT_TILES; ++t) {
    const int j0 = t * JT;
    const char* cur = ktbuf + (t & 1) * KT_BUF_BYTES;

    // -- fire DMA for tile t+1 into the alternate buffer (freed by barrier)
    if (wave == 0 && t + 1 < NT_TILES) issue_tile_load(t + 1, (t + 1) & 1);

    // -- S = Q @ K^T : two 16x16 column tiles, K=256 in 8 WMMA steps
    v8f s0 = zero8(), s1 = zero8();
#pragma unroll
    for (int kk = 0; kk < 8; ++kk) {
      Frag b0, b1;
      const char* pb0 = cur + l16 * 512 + kk * 64 + bsel;
      const char* pb1 = cur + (16 + l16) * 512 + kk * 64 + bsel;
      b0.i4[0] = *(const int4*)pb0; b0.i4[1] = *(const int4*)(pb0 + 16);
      b1.i4[0] = *(const int4*)pb1; b1.i4[1] = *(const int4*)(pb1 + 16);
      s0 = __builtin_amdgcn_wmma_f32_16x16x32_bf16(false, qa[kk].bf, false,
                                                   b0.bf, (short)0, s0, false, false);
      s1 = __builtin_amdgcn_wmma_f32_16x16x32_bf16(false, qa[kk].bf, false,
                                                   b1.bf, (short)0, s1, false, false);
    }

    // -- online softmax update (row r lives in a 16-lane half)
#pragma unroll
    for (int r = 0; r < 8; ++r) {
      float v = fmaxf(s0[r], s1[r]);
      v = fmaxf(v, __shfl_xor(v, 1));
      v = fmaxf(v, __shfl_xor(v, 2));
      v = fmaxf(v, __shfl_xor(v, 4));
      v = fmaxf(v, __shfl_xor(v, 8));
      float mnew  = fmaxf(mrow[r], v);
      float scale = __expf(mrow[r] - mnew);
      mrow[r] = mnew;
      lsum[r] *= scale;
#pragma unroll
      for (int f = 0; f < 16; ++f) acc[f][r] *= scale;
      float p0 = __expf(s0[r] - mnew);
      float p1 = __expf(s1[r] - mnew);
      float ps = p0 + p1;
      ps += __shfl_xor(ps, 1);
      ps += __shfl_xor(ps, 2);
      ps += __shfl_xor(ps, 4);
      ps += __shfl_xor(ps, 8);
      lsum[r] += ps;
      // stage P (bf16) for C->A fragment re-layout
      int row = r + 8 * lh;
      pw[row * 32 + l16]      = f2bf(p0);
      pw[row * 32 + 16 + l16] = f2bf(p1);
    }

    asm volatile("s_wait_dscnt 0" ::: "memory");
    Frag pa;
    {
      const char* pp = pstg + wave * 1024 + l16 * 64 + asel;
      pa.i4[0] = *(const int4*)pp;
      pa.i4[1] = *(const int4*)(pp + 16);
    }

    // -- acc += P @ V : B frags straight from L2-resident global xT[b][d][n]
#pragma unroll
    for (int nt = 0; nt < 16; ++nt) {
      Frag vb;
      const char* pv = (const char*)xT +
          (((size_t)bIdx * Dc + nt * 16 + l16) * Nc + j0) * 2 + bsel;
      vb.i4[0] = *(const int4*)pv;
      vb.i4[1] = *(const int4*)(pv + 16);
      acc[nt] = __builtin_amdgcn_wmma_f32_16x16x32_bf16(false, pa.bf, false,
                                                        vb.bf, (short)0, acc[nt],
                                                        false, false);
    }

    // -- tile t+1 DMA must have landed before anyone reads it next iteration
    if (wave == 0) __builtin_amdgcn_s_wait_tensorcnt(0);
    __syncthreads();
  }

  // ---- epilogue: out = [agg | x] @ Wcat^T  (K = 512, two passes of 256)
  float inv[8];
#pragma unroll
  for (int r = 0; r < 8; ++r) inv[r] = 1.0f / lsum[r];

  char* est = estg + wave * 16 * 512;          // per-wave 16x256 bf16 stage
#pragma unroll
  for (int nt = 0; nt < 16; ++nt) {
#pragma unroll
    for (int r = 0; r < 8; ++r) {
      int row = r + 8 * lh;
      ((unsigned short*)est)[row * 256 + nt * 16 + l16] = f2bf(acc[nt][r] * inv[r]);
    }
  }

  v8f c[16];
#pragma unroll
  for (int i = 0; i < 16; ++i) c[i] = zero8();

#pragma unroll 1
  for (int pass = 0; pass < 2; ++pass) {
    if (pass == 1) {
      // re-stage raw x rows for the residual branch (wave-private region)
      asm volatile("s_wait_dscnt 0" ::: "memory");
      const int4* xr =
          (const int4*)(xbf + ((size_t)bIdx * Nc + i0 + wave * 16) * Dc);
#pragma unroll
      for (int rr = 0; rr < 16; ++rr)
        ((int4*)est)[rr * 32 + lane] = xr[rr * 32 + lane];
      asm volatile("s_wait_dscnt 0" ::: "memory");
    }
#pragma unroll 1
    for (int kk = 0; kk < 8; ++kk) {
      Frag a;
      const char* pa_ = est + l16 * 512 + kk * 64 + asel;
      a.i4[0] = *(const int4*)pa_;
      a.i4[1] = *(const int4*)(pa_ + 16);
#pragma unroll
      for (int nt = 0; nt < 16; ++nt) {
        Frag wb;
        const char* pwv = (const char*)wcat +
                          (size_t)(nt * 16 + l16) * 1024 +
                          (pass * 256 + kk * 32) * 2 + bsel;
        wb.i4[0] = *(const int4*)pwv;
        wb.i4[1] = *(const int4*)(pwv + 16);
        c[nt] = __builtin_amdgcn_wmma_f32_16x16x32_bf16(false, a.bf, false,
                                                        wb.bf, (short)0, c[nt],
                                                        false, false);
      }
    }
  }

  // ---- write pre-BN output + per-channel sum/sumsq partials
  float* orow = out_pre + ((size_t)bIdx * Nc + i0 + wave * 16) * Dc;
#pragma unroll 1
  for (int nt = 0; nt < 16; ++nt) {
    float s = 0.0f, s2 = 0.0f;
#pragma unroll
    for (int r = 0; r < 8; ++r) {
      float v = c[nt][r];
      int row = r + 8 * lh;
      orow[(size_t)row * Dc + nt * 16 + l16] = v;
      s  += v;
      s2 += v * v;
    }
    s  += __shfl_xor(s, 16);
    s2 += __shfl_xor(s2, 16);
    if (lane < 16) {
      atomicAdd(&stats[nt * 16 + l16], s);
      atomicAdd(&stats[256 + nt * 16 + l16], s2);
    }
  }
}

// ---------------------------------------------------------------------------
// Kernel 4: BatchNorm (biased training stats) + SELU, float4 vectorized.
// ---------------------------------------------------------------------------
__global__ void gat_bnselu_kernel(const float* __restrict__ out_pre,
                                  const float* __restrict__ stats,
                                  const float* __restrict__ gamma,
                                  const float* __restrict__ beta,
                                  float* __restrict__ out) {
  const float invM = 1.0f / (float)(Bc * Nc);
  size_t idx = ((size_t)blockIdx.x * blockDim.x + threadIdx.x) * 4;
  if (idx >= (size_t)Bc * Nc * Dc) return;
  float4 v = *(const float4*)(out_pre + idx);
  float vv[4] = {v.x, v.y, v.z, v.w};
  int c0 = (int)(idx & (Dc - 1));
  float o[4];
#pragma unroll
  for (int e = 0; e < 4; ++e) {
    int ch = c0 + e;
    float mean = stats[ch] * invM;
    float var  = stats[256 + ch] * invM - mean * mean;
    float xn = (vv[e] - mean) * rsqrtf(var + 1e-5f) * gamma[ch] + beta[ch];
    float xc = fminf(fmaxf(xn, -10.0f), 10.0f);
    o[e] = 1.0507009873554805f *
           (xn > 0.0f ? xn : 1.6732632423543772f * (__expf(xc) - 1.0f));
  }
  *(float4*)(out + idx) = make_float4(o[0], o[1], o[2], o[3]);
}

// ---------------------------------------------------------------------------
extern "C" void kernel_launch(void* const* d_in, const int* in_sizes, int n_in,
                              void* d_out, int out_size, void* d_ws, size_t ws_size,
                              hipStream_t stream) {
  const float* x     = (const float*)d_in[0];
  const float* w_map = (const float*)d_in[1];
  const float* w_att = (const float*)d_in[2];
  const float* w_res = (const float*)d_in[3];
  const float* gamma = (const float*)d_in[4];
  const float* beta  = (const float*)d_in[5];
  float* out = (float*)d_out;

  // workspace layout
  char* ws = (char*)d_ws;
  const size_t nElem = (size_t)Bc * Nc * Dc;           // 4,194,304
  unsigned short* xbf  = (unsigned short*)(ws);                       // 8 MB
  unsigned short* qbf  = (unsigned short*)(ws + nElem * 2);           // 8 MB
  unsigned short* xT   = (unsigned short*)(ws + nElem * 4);           // 8 MB
  unsigned short* wcat = (unsigned short*)(ws + nElem * 6);           // 256 KB
  float* out_pre = (float*)(ws + nElem * 6 + Dc * 512 * 2);           // 16 MB
  float* stats   = (float*)(ws + nElem * 6 + Dc * 512 * 2 + nElem * 4); // 2 KB

  const int threads = 256;
  const int prepBlocks = (int)((nElem + threads - 1) / threads);      // 16384
  gat_prep_kernel<<<prepBlocks, threads, 0, stream>>>(
      x, w_map, w_att, w_res, xbf, qbf, wcat, stats);

  const int trBlocks = Bc * (Nc / 32) * (Dc / 32);                    // 4096
  gat_transpose_kernel<<<trBlocks, threads, 0, stream>>>(x, xT);

  const int attnBlocks = Bc * (Nc / ROWS_PER_WG);                     // 128
  gat_attn_kernel<<<attnBlocks, threads, SMEM_BYTES, stream>>>(
      xbf, qbf, xT, wcat, out_pre, stats);

  const int bnBlocks = (int)(nElem / 4 / threads);                    // 4096
  gat_bnselu_kernel<<<bnBlocks, threads, 0, stream>>>(
      out_pre, stats, gamma, beta, out);
}
